// LSTM_20409684590858
// MI455X (gfx1250) — matrix operand order repaired and checked by reading, not compiled, not run on hardware
//
#include <hip/hip_runtime.h>

// ---------------------------------------------------------------------------
// LSTM on gfx1250: 1024 sequential steps, each a fused
//   gates = x_t @ Ww^T + h @ Uw^T (+bias)  ->  cell update
// using v_wmma_f32_16x16x32_bf16 with f32 accumulation.
// ---------------------------------------------------------------------------

typedef __attribute__((ext_vector_type(16))) __bf16 v16bf;
typedef __attribute__((ext_vector_type(8)))  float  v8f;

#define B_SZ   64
#define T_SZ   512
#define I_SZ   512
#define H_SZ   1024
#define G_SZ   4096            // 4*H
#define STEPS  1024            // T * NUM_LAYERS

// fp32 -> bf16 (round to nearest even)
__device__ __forceinline__ unsigned short f2bf(float f) {
    unsigned int u = __float_as_uint(f);
    u += 0x7FFFu + ((u >> 16) & 1u);
    return (unsigned short)(u >> 16);
}

union FragU { uint4 q[2]; v16bf v; };

// A-matrix 16x32 bf16 fragment (ISA 7.12.2): lane holds row m = lane%16,
// elements 0..7  = K[kbase + 8*half + 0..7]
// elements 8..15 = K[kbase + 16 + 8*half + 0..7]
__device__ __forceinline__ v16bf load_frag_A(const unsigned short* rowk, int half) {
    FragU u;
    u.q[0] = *(const uint4*)(rowk + 8 * half);
    u.q[1] = *(const uint4*)(rowk + 16 + 8 * half);
    return u.v;
}

// B-matrix 32x16 bf16 fragment: lane holds column n = lane%16,
// elements 0..7  = K[kbase + 8*half + 0..7]
// elements 8..15 = K[kbase + 16 + 8*half + 0..7]
__device__ __forceinline__ v16bf load_frag_B(const unsigned short* rowk, int half) {
    FragU u;
    u.q[0] = *(const uint4*)(rowk + 8 * half);
    u.q[1] = *(const uint4*)(rowk + 16 + 8 * half);
    return u.v;
}

// ---------------------------------------------------------------------------
// Init: convert x/Ww/Uw to bf16, fold biases, zero state.
// ---------------------------------------------------------------------------
__global__ void lstm_init(const float* __restrict__ x,  const float* __restrict__ Ww,
                          const float* __restrict__ bw, const float* __restrict__ Uw,
                          const float* __restrict__ bu,
                          unsigned short* __restrict__ x_bf,
                          unsigned short* __restrict__ Ww_bf,
                          unsigned short* __restrict__ Uw_bf,
                          float* __restrict__ bwu,
                          unsigned short* __restrict__ h_bf,
                          float* __restrict__ c_st) {
    size_t i = (size_t)blockIdx.x * blockDim.x + threadIdx.x;
    if (i < (size_t)B_SZ * T_SZ * I_SZ) x_bf[i]  = f2bf(x[i]);
    if (i < (size_t)G_SZ * I_SZ)        Ww_bf[i] = f2bf(Ww[i]);
    if (i < (size_t)G_SZ * H_SZ)        Uw_bf[i] = f2bf(Uw[i]);
    if (i < G_SZ)                       bwu[i]   = bw[i] + bu[i];
    if (i < (size_t)B_SZ * H_SZ) { h_bf[i] = 0; c_st[i] = 0.0f; }
}

// ---------------------------------------------------------------------------
// One recurrence step. Grid: 64 WGs (16 hidden columns each) x 512 threads
// (16 waves). Wave w: gate q=w/4 (i,f,g,o), batch tile m0=(w%4)*16; computes
// one 16x16 f32 tile over K=512 (x GEMM) + K=1024 (h GEMM) chained WMMAs.
// Gate tiles are combined in LDS for the elementwise cell update.
// ---------------------------------------------------------------------------
__global__ __launch_bounds__(512)
void lstm_step(const unsigned short* __restrict__ x_bf,   // [B][T][I]
               const unsigned short* __restrict__ Ww_bf,  // [4H][I]
               const unsigned short* __restrict__ Uw_bf,  // [4H][H]
               const float* __restrict__ bwu,             // [4H]
               unsigned short* __restrict__ h_bf,         // [B][H]
               float* __restrict__ c_st,                  // [B][H]
               float* __restrict__ out,                   // [2][STEPS][B][H]
               int s) {
    __shared__ float lds[4][B_SZ][16];

    const int tid  = threadIdx.x;
    const int lane = tid & 31;
    const int wv   = tid >> 5;           // 0..15
    const int q    = wv >> 2;            // gate: 0=i 1=f 2=g 3=o
    const int m0   = (wv & 3) << 4;      // batch tile base: 0,16,32,48
    const int half = lane >> 4;
    const int ln   = lane & 15;
    const int h0   = blockIdx.x << 4;    // hidden-column base
    const int t    = s >> 1;             // same x_t for both layer applications

    const int n = q * H_SZ + h0 + ln;    // this lane's gate column (B operand)
    const int m = m0 + ln;               // this lane's batch row   (A operand)

    v8f acc = {};

    // ---- x_t @ Ww^T, K = 512 ----
    const unsigned short* arow = x_bf + (size_t)m * (T_SZ * I_SZ) + (size_t)t * I_SZ;
    const unsigned short* brow = Ww_bf + (size_t)n * I_SZ;
    #pragma unroll 4
    for (int k = 0; k < I_SZ; k += 32) {
        v16bf a = load_frag_A(arow + k, half);
        v16bf b = load_frag_B(brow + k, half);
        acc = __builtin_amdgcn_wmma_f32_16x16x32_bf16(false, a, false, b,
                                                      (short)0, acc, false, false);
    }

    // ---- h @ Uw^T, K = 1024 ----
    const unsigned short* arow2 = h_bf + (size_t)m * H_SZ;
    const unsigned short* brow2 = Uw_bf + (size_t)n * H_SZ;
    #pragma unroll 4
    for (int k = 0; k < H_SZ; k += 32) {
        v16bf a = load_frag_A(arow2 + k, half);
        v16bf b = load_frag_B(brow2 + k, half);
        acc = __builtin_amdgcn_wmma_f32_16x16x32_bf16(false, a, false, b,
                                                      (short)0, acc, false, false);
    }

    // D layout: reg r holds M = m0 + r + 8*half, N = ln (ISA 7.12.2)
    #pragma unroll
    for (int r = 0; r < 8; ++r)
        lds[q][m0 + r + 8 * half][ln] = acc[r];
    __syncthreads();

    // ---- elementwise cell update for this WG's 16 hidden columns ----
    for (int e = tid; e < B_SZ * 16; e += 512) {
        const int b  = e >> 4;
        const int hl = e & 15;
        const int h  = h0 + hl;
        float ig = lds[0][b][hl] + bwu[h];
        float fg = lds[1][b][hl] + bwu[H_SZ + h];
        float gg = lds[2][b][hl] + bwu[2 * H_SZ + h];
        float og = lds[3][b][hl] + bwu[3 * H_SZ + h];
        ig = 1.0f / (1.0f + __expf(-ig));
        fg = 1.0f / (1.0f + __expf(-fg));
        og = 1.0f / (1.0f + __expf(-og));
        gg = tanhf(gg);
        const size_t idx = (size_t)b * H_SZ + h;
        const float c_new = c_st[idx] * fg + ig * gg;
        const float h_new = og * tanhf(c_new);
        c_st[idx] = c_new;                         // f32 carry state
        h_bf[idx] = f2bf(h_new);                   // bf16 GEMM state
        const size_t o = (size_t)s * (B_SZ * H_SZ) + idx;
        out[o] = h_new;                                        // hs
        out[(size_t)STEPS * B_SZ * H_SZ + o] = c_new;          // cs
    }
}

// ---------------------------------------------------------------------------
extern "C" void kernel_launch(void* const* d_in, const int* in_sizes, int n_in,
                              void* d_out, int out_size, void* d_ws, size_t ws_size,
                              hipStream_t stream) {
    const float* x  = (const float*)d_in[0];
    const float* Ww = (const float*)d_in[1];
    const float* bw = (const float*)d_in[2];
    const float* Uw = (const float*)d_in[3];
    const float* bu = (const float*)d_in[4];
    float* out = (float*)d_out;

    // workspace layout (bytes)
    const size_t OFF_XBF  = 0;                          // 33,554,432
    const size_t OFF_WWBF = 33554432;                   //  4,194,304
    const size_t OFF_UWBF = 37748736;                   //  8,388,608
    const size_t OFF_BWU  = 46137344;                   //     16,384
    const size_t OFF_HBF  = 46153728;                   //    131,072
    const size_t OFF_C    = 46284800;                   //    262,144
    const size_t NEED     = 46546944;
    if (ws_size < NEED) return;                         // avoid OOB on tiny ws

    char* ws = (char*)d_ws;
    unsigned short* x_bf  = (unsigned short*)(ws + OFF_XBF);
    unsigned short* Ww_bf = (unsigned short*)(ws + OFF_WWBF);
    unsigned short* Uw_bf = (unsigned short*)(ws + OFF_UWBF);
    float*          bwu   = (float*)(ws + OFF_BWU);
    unsigned short* h_bf  = (unsigned short*)(ws + OFF_HBF);
    float*          c_st  = (float*)(ws + OFF_C);

    // init: one thread per x element (largest array)
    lstm_init<<<(B_SZ * T_SZ * I_SZ) / 256, 256, 0, stream>>>(
        x, Ww, bw, Uw, bu, x_bf, Ww_bf, Uw_bf, bwu, h_bf, c_st);

    // sequential recurrence: stream ordering is the cross-step barrier
    for (int s = 0; s < STEPS; ++s)
        lstm_step<<<H_SZ / 16, 512, 0, stream>>>(
            x_bf, Ww_bf, Uw_bf, bwu, h_bf, c_st, out, s);
}
